// EdgeGCN_77962246357191
// MI455X (gfx1250) — matrix operand order repaired and checked by reading, not compiled, not run on hardware
//
#include <hip/hip_runtime.h>

typedef __attribute__((ext_vector_type(16))) _Float16 v16h;
typedef __attribute__((ext_vector_type(8)))  float    v8f;

#define N_NODES  50000
#define N_EDGES  800000
#define NODE_IN  128
#define EDGE_IN  16
#define KDIM     (NODE_IN + EDGE_IN)   // 144
#define KPAD     160                   // 5 * 32
#define KTILES   5
#define NCOLS    128
#define NTILES   8                     // 128 / 16
#define ROWTILES (N_NODES / 16)        // 3125 (exact)

// ---------------------------------------------------------------------------
// zero a float buffer (grid-stride)
__global__ void zero_f32(float* __restrict__ p, int n) {
  int i = blockIdx.x * blockDim.x + threadIdx.x;
  int stride = gridDim.x * blockDim.x;
  for (; i < n; i += stride) p[i] = 0.0f;
}

// ---------------------------------------------------------------------------
// Pass 1 scatter: SX[dst] += x[src] (128 f32), SE[dst] += edge_attr[e] (16 f32)
// One wave (32 lanes) per edge. Edge id forced scalar so src/dst use SMEM
// loads; gathered row read as one b128 per lane. Atomics resolve in L2
// (all live buffers total ~70MB < 192MB L2).
__global__ void __launch_bounds__(256)
scatter_x_e(const float* __restrict__ x,
            const long long* __restrict__ ei,
            const float* __restrict__ ea,
            float* __restrict__ SX,
            float* __restrict__ SE) {
  int lane = threadIdx.x & 31;
  int e = __builtin_amdgcn_readfirstlane(blockIdx.x * 8 + (threadIdx.x >> 5));
  if (e >= N_EDGES) return;
  int src = (int)ei[e];
  int dst = (int)ei[N_EDGES + e];

  const float4* xs = (const float4*)(x + (long long)src * NODE_IN);
  float4 v = xs[lane];                                   // global_load_b128
  float* sxd = SX + (long long)dst * NODE_IN + lane * 4;
  atomicAdd(&sxd[0], v.x);
  atomicAdd(&sxd[1], v.y);
  atomicAdd(&sxd[2], v.z);
  atomicAdd(&sxd[3], v.w);

  if (lane < EDGE_IN)
    atomicAdd(&SE[(long long)dst * EDGE_IN + lane],
              ea[(long long)e * EDGE_IN + lane]);
}

// Pass 2 scatter: SH[dst] += H[src] (128 f32)
__global__ void __launch_bounds__(256)
scatter_h(const float* __restrict__ H,
          const long long* __restrict__ ei,
          float* __restrict__ SH) {
  int lane = threadIdx.x & 31;
  int e = __builtin_amdgcn_readfirstlane(blockIdx.x * 8 + (threadIdx.x >> 5));
  if (e >= N_EDGES) return;
  int src = (int)ei[e];
  int dst = (int)ei[N_EDGES + e];

  const float4* hs = (const float4*)(H + (long long)src * NODE_IN);
  float4 v = hs[lane];
  float* shd = SH + (long long)dst * NODE_IN + lane * 4;
  atomicAdd(&shd[0], v.x);
  atomicAdd(&shd[1], v.y);
  atomicAdd(&shd[2], v.z);
  atomicAdd(&shd[3], v.w);
}

// ---------------------------------------------------------------------------
// Pack A fragments (f16) in WMMA A layout (16-bit A 16x32, ISA 7.12.2):
//   lane<16 : halves 0..7 -> K=kb+0..7,  halves 8..15 -> K=kb+16..23  (kb=t*32)
//   lane>=16: halves 0..7 -> K=kb+0..7,  halves 8..15 -> K=kb+16..23  (kb=t*32+8)
// One thread per fragment-lane: writes 16 contiguous halves (32B, b128x2).
// K-runs are 8-aligned so they never straddle the 128 / 144 boundaries.
__global__ void pack_A(const float* __restrict__ S128,   // [N_NODES][128]
                       const float* __restrict__ S16,    // [N_NODES][16]
                       _Float16* __restrict__ Apack) {
  int id = blockIdx.x * blockDim.x + threadIdx.x;        // < N_NODES*KPAD/16
  if (id >= N_NODES * (KPAD / 16)) return;
  int l  = id & 31;
  int rt = id >> 5;
  int t  = rt % KTILES;
  int r  = rt / KTILES;
  int M     = r * 16 + (l & 15);
  int kbase = t * 32 + ((l < 16) ? 0 : 8);

  const float* row128 = S128 + (long long)M * NODE_IN;
  const float* row16  = S16  + (long long)M * EDGE_IN;

  v16h out;
#pragma unroll
  for (int h = 0; h < 16; ++h) {
    int K = kbase + ((h < 8) ? h : h + 8);
    float v = 0.0f;
    if (K < NODE_IN)   v = row128[K];
    else if (K < KDIM) v = row16[K - NODE_IN];
    out[h] = (_Float16)v;
  }
  ((v16h*)Apack)[id] = out;
}

// Pack both weight matrices (144x128 f32, row-major) into WMMA B layout
// (16-bit B 32x16): lane<16 -> N=l, K=t*32+(h<8?h:h+8); lane>=16 -> N=l-16, K=t*32+8+(h<8?h:h+8).
__global__ void pack_W(const float* __restrict__ W1,
                       const float* __restrict__ W2,
                       _Float16* __restrict__ W1p,
                       _Float16* __restrict__ W2p) {
  const int PER = NTILES * KTILES * 32 * 16;        // 20480
  int id = blockIdx.x * blockDim.x + threadIdx.x;
  if (id >= 2 * PER) return;
  const float* W  = (id < PER) ? W1  : W2;
  _Float16*    Wp = (id < PER) ? W1p : W2p;
  int lid  = (id < PER) ? id : id - PER;
  int h    = lid & 15;
  int frag = lid >> 4;
  int l    = frag & 31;
  int jt   = frag >> 5;
  int t    = jt % KTILES;
  int j    = jt / KTILES;
  int N = j * 16 + (l & 15);
  int K = t * 32 + ((l < 16) ? 0 : 8) + ((h < 8) ? h : h + 8);
  float v = (K < KDIM) ? W[(long long)K * NCOLS + N] : 0.0f;
  Wp[lid] = (_Float16)v;
}

// ---------------------------------------------------------------------------
// GEMM: out[50000x128] = act( Apack[50000x160] @ Wp[160x128] + bias )
// 4 waves/block, 1 wave = one 16-row strip. 8 N-tiles x 5 K-chunks = 40 WMMAs.
__global__ void __launch_bounds__(128)
wmma_gemm(const _Float16* __restrict__ Apack,
          const _Float16* __restrict__ Wp,
          const float* __restrict__ bias,
          float* __restrict__ out,
          int do_relu) {
  int wave = threadIdx.x >> 5;
  int lane = threadIdx.x & 31;
  int r = blockIdx.x * 4 + wave;
  if (r >= ROWTILES) return;                 // wave-uniform exit, EXEC stays full

  const v16h* A = (const v16h*)Apack;
  const v16h* B = (const v16h*)Wp;

  v16h a[KTILES];
#pragma unroll
  for (int t = 0; t < KTILES; ++t)
    a[t] = A[((long long)r * KTILES + t) * 32 + lane];

  int col_lo = lane & 15;
  int mbase  = r * 16 + ((lane >> 4) << 3);  // +8 for lanes 16..31

#pragma unroll
  for (int j = 0; j < NTILES; ++j) {
    float bv = bias[j * 16 + col_lo];        // same N for all 8 acc VGPRs
    v8f acc;
#pragma unroll
    for (int v = 0; v < 8; ++v) acc[v] = bv;
#pragma unroll
    for (int t = 0; t < KTILES; ++t) {
      v16h b = B[(j * KTILES + t) * 32 + lane];
      acc = __builtin_amdgcn_wmma_f32_16x16x32_f16(
          /*neg_a=*/false, a[t], /*neg_b=*/false, b,
          /*c_mod=*/(short)0, acc, /*reuse_a=*/false, /*reuse_b=*/false);
    }
#pragma unroll
    for (int v = 0; v < 8; ++v) {
      float val = acc[v];
      if (do_relu) val = fmaxf(val, 0.0f);
      out[(long long)(mbase + v) * NCOLS + j * 16 + col_lo] = val;
    }
  }
}

// ---------------------------------------------------------------------------
extern "C" void kernel_launch(void* const* d_in, const int* in_sizes, int n_in,
                              void* d_out, int out_size, void* d_ws, size_t ws_size,
                              hipStream_t stream) {
  const float*     x  = (const float*)d_in[0];
  const long long* ei = (const long long*)d_in[1];   // int64 edge_index (2,E)
  const float*     ea = (const float*)d_in[2];
  const float*     W1 = (const float*)d_in[3];
  const float*     b1 = (const float*)d_in[4];
  const float*     W2 = (const float*)d_in[5];
  const float*     b2 = (const float*)d_in[6];
  float* out = (float*)d_out;

  char* ws = (char*)d_ws;
  size_t off = 0;
  auto take = [&](size_t bytes) -> void* {
    void* p = ws + off;
    off = (off + bytes + 255) & ~(size_t)255;
    return p;
  };
  float*    SX    = (float*)take((size_t)N_NODES * NODE_IN * 4);   // 25.6 MB (reused as SH)
  float*    SE    = (float*)take((size_t)N_NODES * EDGE_IN * 4);   //  3.2 MB
  float*    H     = (float*)take((size_t)N_NODES * NODE_IN * 4);   // 25.6 MB
  _Float16* Apack = (_Float16*)take((size_t)N_NODES * KPAD * 2);   // 16.0 MB
  _Float16* W1p   = (_Float16*)take((size_t)20480 * 2);
  _Float16* W2p   = (_Float16*)take((size_t)20480 * 2);

  const int ZB = 2048, TB = 256;
  int scatterBlocks = N_EDGES / 8;                           // 100000 (exact)
  int packABlocks   = (N_NODES * (KPAD / 16) + TB - 1) / TB; // 1954
  int gemmBlocks    = (ROWTILES + 3) / 4;                    // 782

  // Layer 1: aggregate -> pack -> GEMM(relu)
  zero_f32<<<ZB, TB, 0, stream>>>(SX, N_NODES * NODE_IN);
  zero_f32<<<512, TB, 0, stream>>>(SE, N_NODES * EDGE_IN);
  pack_W<<<(2 * 20480 + TB - 1) / TB, TB, 0, stream>>>(W1, W2, W1p, W2p);
  scatter_x_e<<<scatterBlocks, TB, 0, stream>>>(x, ei, ea, SX, SE);
  pack_A<<<packABlocks, TB, 0, stream>>>(SX, SE, Apack);
  wmma_gemm<<<gemmBlocks, 128, 0, stream>>>(Apack, W1p, b1, H, 1);

  // Layer 2: aggregate H -> pack (SE reused) -> GEMM (no relu) -> d_out
  zero_f32<<<ZB, TB, 0, stream>>>(SX, N_NODES * NODE_IN);    // SX reused as SH
  scatter_h<<<scatterBlocks, TB, 0, stream>>>(H, ei, SX);
  pack_A<<<packABlocks, TB, 0, stream>>>(SX, SE, Apack);
  wmma_gemm<<<gemmBlocks, 128, 0, stream>>>(Apack, W2p, b2, out, 0);
}
